// Attention_2559800508752
// MI455X (gfx1250) — matrix operand order repaired and checked by reading, not compiled, not run on hardware
//
#include <hip/hip_runtime.h>
#include <hip/hip_bf16.h>

typedef __bf16 bf16;
typedef bf16  bf16x8  __attribute__((ext_vector_type(8)));
typedef bf16  bf16x16 __attribute__((ext_vector_type(16)));
typedef float f32x8   __attribute__((ext_vector_type(8)));

#define BATCH   4
#define SEQ     2048
#define DMODEL  1024
#define NHEADS  16
#define DHEAD   64
#define MROWS   (BATCH*SEQ)   /* 8192 */
#define IGNORE_VAL (-1e5f)

static __device__ __forceinline__ f32x8
wmma_bf16(bf16x16 a, bf16x16 b, f32x8 c) {
  return __builtin_amdgcn_wmma_f32_16x16x32_bf16(false, a, false, b,
                                                 (short)0, c, false, false);
}

// Pack an A-fragment: two contiguous 8-element chunks (k = half*16 .. +7 and
// k = half*16 + 8 .. +7), i.e. contiguous k = 16*half + s, same order as the B loads.
static __device__ __forceinline__ bf16x16
pack16(const bf16* lo, const bf16* hi) {
  bf16x8 l = *(const bf16x8*)lo;
  bf16x8 h = *(const bf16x8*)hi;
  bf16x16 r;
#pragma unroll
  for (int j = 0; j < 8; ++j) { r[j] = l[j]; r[j+8] = h[j]; }
  return r;
}

// ---------------- converters ----------------

__global__ void k_f32_to_bf16(const float* __restrict__ in,
                              bf16* __restrict__ out, int n) {
  int i = blockIdx.x * blockDim.x + threadIdx.x;
  int stride = gridDim.x * blockDim.x;
  for (; i < n; i += stride) out[i] = (bf16)in[i];
}

// W_{Q,K,V}[H, DMODEL, DHEAD] f32  ->  Bt[c = h*64+d][m] bf16   (c-major)
__global__ void k_conv_wqkv(const float* __restrict__ w, bf16* __restrict__ out) {
  int i = blockIdx.x * blockDim.x + threadIdx.x;
  int stride = gridDim.x * blockDim.x;
  const int n = DMODEL * NHEADS * DHEAD;
  for (; i < n; i += stride) {
    int c = i >> 10;      // column (h*64+d)
    int m = i & 1023;     // k index (d_model)
    int h = c >> 6, d = c & 63;
    out[i] = (bf16)w[((size_t)h * DMODEL + m) * DHEAD + d];
  }
}

// W_O[H, DHEAD, DMODEL] f32 (flat [1024][1024]) -> Bt[m][h*64+d] bf16
__global__ void k_conv_wo(const float* __restrict__ w, bf16* __restrict__ out) {
  int i = blockIdx.x * blockDim.x + threadIdx.x;
  int stride = gridDim.x * blockDim.x;
  const int n = DMODEL * DMODEL;
  for (; i < n; i += stride) {
    int m = i >> 10;      // output column (d_model)
    int c = i & 1023;     // k index (h*64+d)
    out[i] = (bf16)w[(size_t)c * DMODEL + m];
  }
}

// ---------------- generic bf16 WMMA GEMM ----------------
// C[M x 1024] = A[M x 1024] * B  (B stored transposed: Bt[n][k]) + bias[n]
// Wave tile: 32 rows x 64 cols (2 A-frags share 4 B-frags -> 8 WMMA / 6 loads).
// Block: 4 waves = 128 rows x 64 cols.
// mode 0: bf16 row-major out       (Q, K)
// mode 1: bf16 out transposed to Vt[(b*H+h)*64+d][s]   (V)
// mode 2: f32 row-major out        (final projection -> d_out)
__global__ void k_gemm(const bf16* __restrict__ A, const bf16* __restrict__ Bt,
                       const float* __restrict__ bias,
                       bf16* __restrict__ outB, float* __restrict__ outF,
                       int mode) {
  const int K = DMODEL;
  int lane = threadIdx.x & 31;
  int w    = threadIdx.x >> 5;
  int ln   = lane & 15;
  int half = lane >> 4;
  int m0 = blockIdx.x * 128 + w * 32;  // 32 rows per wave
  int n0 = blockIdx.y * 64;            // 64 cols per wave (4 tiles)

  f32x8 acc[2][4];
#pragma unroll
  for (int u = 0; u < 2; ++u)
#pragma unroll
    for (int t = 0; t < 4; ++t)
#pragma unroll
      for (int j = 0; j < 8; ++j) acc[u][t][j] = 0.0f;

  const bf16* arow0 = A + (size_t)(m0 + ln) * K;
  const bf16* arow1 = A + (size_t)(m0 + 16 + ln) * K;
  for (int ks = 0; ks < K; ks += 32) {
    bf16x16 a0 = pack16(arow0 + ks + half * 16, arow0 + ks + half * 16 + 8);
    bf16x16 a1 = pack16(arow1 + ks + half * 16, arow1 + ks + half * 16 + 8);
#pragma unroll
    for (int t = 0; t < 4; ++t) {
      const bf16* bp = Bt + (size_t)(n0 + t * 16 + ln) * K + ks + half * 16;
      bf16x16 b = *(const bf16x16*)bp;
      acc[0][t] = wmma_bf16(a0, b, acc[0][t]);
      acc[1][t] = wmma_bf16(a1, b, acc[1][t]);
    }
  }

#pragma unroll
  for (int u = 0; u < 2; ++u) {
    int mu = m0 + u * 16;
#pragma unroll
    for (int t = 0; t < 4; ++t) {
      int col = n0 + t * 16 + ln;
      float bias_v = bias[col];
      if (mode == 0) {
#pragma unroll
        for (int r = 0; r < 8; ++r) {
          int row = mu + r + 8 * half;
          outB[(size_t)row * DMODEL + col] = (bf16)(acc[u][t][r] + bias_v);
        }
      } else if (mode == 1) {
        int bidx = mu / SEQ;
        int srow = (mu % SEQ) + 8 * half;
        int hh = col >> 6, dd = col & 63;
        bf16x8 v;
#pragma unroll
        for (int r = 0; r < 8; ++r) v[r] = (bf16)(acc[u][t][r] + bias_v);
        *(bf16x8*)(outB + (size_t)((bidx * NHEADS + hh) * DHEAD + dd) * SEQ + srow) = v;
      } else {
#pragma unroll
        for (int r = 0; r < 8; ++r) {
          int row = mu + r + 8 * half;
          outF[(size_t)row * DMODEL + col] = acc[u][t][r] + bias_v;
        }
      }
    }
  }
}

// ---------------- flash attention ----------------
// Q, K: bf16 [b, s, h*64+d] row-major.  Vt: bf16 [(b*H+h)*64+d][s].
// Z out: bf16 [b, s, h*64+d].
__global__ void k_attn(const bf16* __restrict__ Q, const bf16* __restrict__ Kb,
                       const bf16* __restrict__ Vt, bf16* __restrict__ Z) {
  __shared__ bf16 Plds[4][16 * 32];   // per-wave P scratch (C-layout -> A-layout)
  int lane = threadIdx.x & 31;
  int w    = threadIdx.x >> 5;
  int ln   = lane & 15;
  int half = lane >> 4;
  int bh = blockIdx.y;
  int b  = bh >> 4;
  int h  = bh & 15;
  int q0 = blockIdx.x * 64 + w * 16;   // 16 queries per wave

  // Q A-fragments (fixed over key loop): two k-steps over d_head
  const bf16* qrow = Q + (size_t)(b * SEQ + q0 + ln) * DMODEL + h * DHEAD;
  bf16x16 aq[2];
#pragma unroll
  for (int ks = 0; ks < 2; ++ks)
    aq[ks] = pack16(qrow + ks * 32 + half * 16, qrow + ks * 32 + half * 16 + 8);

  f32x8 o[4];
#pragma unroll
  for (int t = 0; t < 4; ++t)
#pragma unroll
    for (int j = 0; j < 8; ++j) o[t][j] = 0.0f;
  float mrow[8], lrow[8];
#pragma unroll
  for (int r = 0; r < 8; ++r) { mrow[r] = -1e30f; lrow[r] = 0.0f; }

  const float inv_sqrt_d = 0.125f;
  int nkb = (q0 + 16 + 31) >> 5;   // key blocks of 32, only up to causal limit

  for (int kb = 0; kb < nkb; ++kb) {
    int key0 = kb * 32;
    // scores: 2 key subtiles x 2 d-steps
    f32x8 s0, s1;
#pragma unroll
    for (int j = 0; j < 8; ++j) { s0[j] = 0.0f; s1[j] = 0.0f; }
#pragma unroll
    for (int ks = 0; ks < 2; ++ks) {
      const bf16* k0p = Kb + (size_t)(b * SEQ + key0 + ln) * DMODEL
                           + h * DHEAD + ks * 32 + half * 16;
      const bf16* k1p = Kb + (size_t)(b * SEQ + key0 + 16 + ln) * DMODEL
                           + h * DHEAD + ks * 32 + half * 16;
      bf16x16 b0 = *(const bf16x16*)k0p;
      bf16x16 b1 = *(const bf16x16*)k1p;
      s0 = wmma_bf16(aq[ks], b0, s0);
      s1 = wmma_bf16(aq[ks], b1, s1);
    }
    // scale + causal mask (reference: scale first, masked value exactly -1e5)
    float p0[8], p1[8], rm[8];
#pragma unroll
    for (int r = 0; r < 8; ++r) {
      int qi = q0 + r + 8 * half;
      float v0 = s0[r] * inv_sqrt_d;
      float v1 = s1[r] * inv_sqrt_d;
      if (key0 + ln > qi)       v0 = IGNORE_VAL;
      if (key0 + 16 + ln > qi)  v1 = IGNORE_VAL;
      p0[r] = v0; p1[r] = v1;
      rm[r] = fmaxf(v0, v1);
    }
    // row max across 16-lane group
#pragma unroll
    for (int off = 1; off < 16; off <<= 1)
#pragma unroll
      for (int r = 0; r < 8; ++r) rm[r] = fmaxf(rm[r], __shfl_xor(rm[r], off, 32));
    float rs[8];
#pragma unroll
    for (int r = 0; r < 8; ++r) {
      float mnew = fmaxf(mrow[r], rm[r]);
      float scale = __expf(mrow[r] - mnew);
      p0[r] = __expf(p0[r] - mnew);
      p1[r] = __expf(p1[r] - mnew);
      rs[r] = p0[r] + p1[r];
      lrow[r] = lrow[r] * scale;
      mrow[r] = mnew;
#pragma unroll
      for (int t = 0; t < 4; ++t) o[t][r] *= scale;
    }
#pragma unroll
    for (int off = 1; off < 16; off <<= 1)
#pragma unroll
      for (int r = 0; r < 8; ++r) rs[r] += __shfl_xor(rs[r], off, 32);
#pragma unroll
    for (int r = 0; r < 8; ++r) lrow[r] += rs[r];

    // P (C-layout) -> LDS row-major [16 x 32] bf16
    bf16* P = Plds[w];
#pragma unroll
    for (int r = 0; r < 8; ++r) {
      int m = r + 8 * half;
      P[m * 32 + ln]      = (bf16)p0[r];
      P[m * 32 + 16 + ln] = (bf16)p1[r];
    }
    asm volatile("s_wait_dscnt 0" ::: "memory");
    // A-fragment of P from LDS
    const bf16* pr = P + ln * 32;
    bf16x16 ap = pack16(pr + half * 16, pr + half * 16 + 8);
    // PV: 4 d-tiles, K = 32 keys
#pragma unroll
    for (int t = 0; t < 4; ++t) {
      const bf16* vp = Vt + (size_t)(bh * DHEAD + t * 16 + ln) * SEQ
                          + key0 + half * 16;
      bf16x16 bv = *(const bf16x16*)vp;
      o[t] = wmma_bf16(ap, bv, o[t]);
    }
  }

  // normalize and emit Z [b, s, h*64+d] bf16
  float rinv[8];
#pragma unroll
  for (int r = 0; r < 8; ++r) rinv[r] = 1.0f / lrow[r];
#pragma unroll
  for (int t = 0; t < 4; ++t)
#pragma unroll
    for (int r = 0; r < 8; ++r) {
      int row = b * SEQ + q0 + r + 8 * half;
      Z[(size_t)row * DMODEL + h * DHEAD + t * 16 + ln] = (bf16)(o[t][r] * rinv[r]);
    }
}

// ---------------- launcher ----------------

extern "C" void kernel_launch(void* const* d_in, const int* in_sizes, int n_in,
                              void* d_out, int out_size, void* d_ws, size_t ws_size,
                              hipStream_t stream) {
  (void)in_sizes; (void)n_in; (void)out_size; (void)ws_size;
  const float* resid = (const float*)d_in[0];
  const float* W_Q   = (const float*)d_in[1];
  const float* W_K   = (const float*)d_in[2];
  const float* W_V   = (const float*)d_in[3];
  const float* W_O   = (const float*)d_in[4];
  const float* b_Q   = (const float*)d_in[5];
  const float* b_K   = (const float*)d_in[6];
  const float* b_V   = (const float*)d_in[7];
  const float* b_O   = (const float*)d_in[8];
  float* out = (float*)d_out;

  char* ws = (char*)d_ws;
  size_t off = 0;
  auto carve = [&](size_t bytes) {
    void* p = ws + off;
    off += (bytes + 255) & ~(size_t)255;
    return p;
  };
  bf16* X   = (bf16*)carve((size_t)MROWS * DMODEL * 2);          // 16 MB
  bf16* WqT = (bf16*)carve((size_t)DMODEL * DMODEL * 2);         //  2 MB
  bf16* WkT = (bf16*)carve((size_t)DMODEL * DMODEL * 2);
  bf16* WvT = (bf16*)carve((size_t)DMODEL * DMODEL * 2);
  bf16* WoT = (bf16*)carve((size_t)DMODEL * DMODEL * 2);
  bf16* Qb  = (bf16*)carve((size_t)MROWS * DMODEL * 2);          // 16 MB
  bf16* Kb  = (bf16*)carve((size_t)MROWS * DMODEL * 2);
  bf16* Vt  = (bf16*)carve((size_t)MROWS * DMODEL * 2);
  bf16* Zb  = (bf16*)carve((size_t)MROWS * DMODEL * 2);

  k_f32_to_bf16<<<2048, 256, 0, stream>>>(resid, X, MROWS * DMODEL);
  k_conv_wqkv<<<1024, 256, 0, stream>>>(W_Q, WqT);
  k_conv_wqkv<<<1024, 256, 0, stream>>>(W_K, WkT);
  k_conv_wqkv<<<1024, 256, 0, stream>>>(W_V, WvT);
  k_conv_wo  <<<1024, 256, 0, stream>>>(W_O, WoT);

  dim3 gG(MROWS / 128, DMODEL / 64);
  k_gemm<<<gG, 128, 0, stream>>>(X, WqT, b_Q, Qb, nullptr, 0);
  k_gemm<<<gG, 128, 0, stream>>>(X, WkT, b_K, Kb, nullptr, 0);
  k_gemm<<<gG, 128, 0, stream>>>(X, WvT, b_V, Vt, nullptr, 1);

  dim3 gA(SEQ / 64, BATCH * NHEADS);
  k_attn<<<gA, 128, 0, stream>>>(Qb, Kb, Vt, Zb);

  k_gemm<<<gG, 128, 0, stream>>>(Zb, WoT, b_O, nullptr, out, 2);
}
